// New_CrfRnnLayer_3d_GPU_2_2_posterior_66949950210612
// MI455X (gfx1250) — compile-verified, matrix operand order checked by reading, not run on hardware
//
#include <hip/hip_runtime.h>

#define Dz 24
#define Hy 24
#define Wx 16
#define Cc 2
#define Nn (Dz*Hy*Wx)          // 9216 voxels

typedef __attribute__((ext_vector_type(16))) _Float16 v16h;
typedef __attribute__((ext_vector_type(2)))  _Float16 h2;
typedef __attribute__((ext_vector_type(8)))  float    v8f;

// 8-byte feature record per voxel: intensity (f32) + q0,q1 (f16x2)
struct __align__(8) Feat { float I; h2 q; };

static __device__ __forceinline__ h2 splat_h(float v) {
    h2 r; r.x = (_Float16)v; r.y = (_Float16)v; return r;
}

// One mean-field iteration: qout = u - compat( w_s * Ks@q/rowsum(Ks) + w_b * Kb@q/rowsum(Kb) )
// Each wave owns one 16-row output tile. Per 32-column j-step it builds two 16x32
// f16 A tiles (spatial & bilateral kernel values) and one 32x16 f16 B tile whose
// columns are [q0 | q1 | ones | 0...]; one WMMA per kernel accumulates both
// filtered classes plus the normalizer. Gaussians are factored per axis:
//   exp(-.5*d2) = e_x[slot] * e_yz[step] * e_I(pair)
// e_x is precomputed (packed f16), e_yz costs 4 exps/step, only e_I is per-pair.
__global__ __launch_bounds__(256) void crf_iter(
    const float* __restrict__ u,    const float* __restrict__ rgb,
    const float* __restrict__ qin,  float* __restrict__ qout,
    const float* __restrict__ sw,   const float* __restrict__ bw,
    const float* __restrict__ cm)
{
    extern __shared__ Feat sF[];    // [Nn]

    const int tid = threadIdx.x;
    for (int j = tid; j < Nn; j += 256) {   // 9216/256 = 36, uniform
        Feat f;
        f.I   = rgb[j];
        f.q.x = (_Float16)qin[2*j + 0];
        f.q.y = (_Float16)qin[2*j + 1];
        sF[j] = f;
    }
    __syncthreads();

    const int lane  = tid & 31;
    const int wave  = tid >> 5;
    const int hi    = lane >> 4;               // which 16-lane half
    const int lm    = lane & 15;
    const int itile = blockIdx.x * 8 + wave;   // 72 blocks * 8 waves = 576 tiles
    const int irow  = itile * 16 + lm;         // A-matrix row owned by this lane

    const float mg = -0.5f / (3.0f * 3.0f);      // -1/(2*theta_gamma^2)
    const float ma = -0.5f / (160.0f * 160.0f);  // -1/(2*theta_alpha^2)
    const float mb = -0.5f / (3.0f * 3.0f);      // -1/(2*theta_beta^2)

    const int   irw = irow >> 4;                 // row index in (z,y) plane
    const float ix  = (float)(irow & 15);
    const float iy  = (float)(irw % Hy);
    const float iz  = (float)(irw / Hy);
    const float iI  = sF[irow].I;

    // Per-slot x factors, packed as f16 pairs: x(j) = (j0+k)&15 = hi*8 + 2p + t
    h2 esx_pk[4], ebx_pk[4];
#pragma unroll
    for (int p = 0; p < 4; ++p) {
        const float dx0 = ix - (float)(hi * 8 + 2 * p);
        const float dx1 = ix - (float)(hi * 8 + 2 * p + 1);
        h2 es, eb;
        es.x = (_Float16)__expf(mg * dx0 * dx0);
        es.y = (_Float16)__expf(mg * dx1 * dx1);
        eb.x = (_Float16)__expf(ma * dx0 * dx0);
        eb.y = (_Float16)__expf(ma * dx1 * dx1);
        esx_pk[p] = es;
        ebx_pk[p] = eb;
    }

    // Per-lane B-column blend masks (column lm carries q0 / q1 / ones)
    const _Float16 bm0 = (lm == 0) ? (_Float16)1.0f : (_Float16)0.0f;
    const _Float16 bm1 = (lm == 1) ? (_Float16)1.0f : (_Float16)0.0f;
    const _Float16 bad = (lm == 2) ? (_Float16)1.0f : (_Float16)0.0f;

    v8f acc_s = {};   // cols: [Ks@q0 | Ks@q1 | rowsum(Ks) | ...]
    v8f acc_b = {};

    // Uniform (y,z) counters for plane-row ry = j0>>4 (advances by 2 per step)
    int y0i = 0, z0i = 0;

    for (int j0 = 0; j0 < Nn; j0 += 32) {
        // The 32 j's of this step live in plane-rows ry0=j0>>4 and ry0+1
        int y1i = y0i + 1, z1i = z0i;
        if (y1i == Hy) { y1i = 0; z1i = z0i + 1; }
        const float dy0 = iy - (float)y0i, dz0 = iz - (float)z0i;
        const float dy1 = iy - (float)y1i, dz1 = iz - (float)z1i;
        const float r0 = dy0 * dy0 + dz0 * dz0;
        const float r1 = dy1 * dy1 + dz1 * dz1;
        // 4 exps per step for the (y,z) factors, splatted to packed f16
        const h2 es0 = splat_h(__expf(mg * r0)), es1 = splat_h(__expf(mg * r1));
        const h2 eb0 = splat_h(__expf(ma * r0)), eb1 = splat_h(__expf(ma * r1));

        v16h A_s, A_b, Bm;
#pragma unroll
        for (int v = 0; v < 8; ++v) {
            // 16-bit A-matrix 16x32 layout: VGPR v, lane-half -> K index
            const int p   = v & 3;                     // x-slot pair 0..3
            const int sel = (v >= 4);                  // upper 16 of K
            const int jb  = j0 + (sel ? 16 : 0) + hi * 8 + 2 * p;

            const Feat fa = sF[jb];                    // ds_load_b64
            const Feat fb = sF[jb + 1];                // ds_load_b64

            // Spatial kernel pair: pure packed f16
            const h2 As = (sel ? es1 : es0) * esx_pk[p];

            // Bilateral: per-pair intensity exp (the only per-element exp)
            const float dIa = iI - fa.I;
            const float dIb = iI - fb.I;
            h2 eI;
            eI.x = (_Float16)__expf(mb * dIa * dIa);
            eI.y = (_Float16)__expf(mb * dIb * dIb);
            const h2 Ab = ((sel ? eb1 : eb0) * ebx_pk[p]) * eI;

            // B tile pair, branchless scalar f16 FMAs (avoids half repacking)
            const _Float16 Bpa = fa.q.x * bm0 + (fa.q.y * bm1 + bad);
            const _Float16 Bpb = fb.q.x * bm0 + (fb.q.y * bm1 + bad);

            A_s[2*v] = As.x;  A_s[2*v + 1] = As.y;
            A_b[2*v] = Ab.x;  A_b[2*v + 1] = Ab.y;
            Bm [2*v] = Bpa;   Bm [2*v + 1] = Bpb;
        }
        // D = A*B + C  (f16 in, f32 accumulate)
        acc_s = __builtin_amdgcn_wmma_f32_16x16x32_f16(
                    false, A_s, false, Bm, (short)0, acc_s, false, false);
        acc_b = __builtin_amdgcn_wmma_f32_16x16x32_f16(
                    false, A_b, false, Bm, (short)0, acc_b, false, false);

        // advance uniform (y,z) by 2 plane-rows
        y0i += 2;
        if (y0i >= Hy) { y0i -= Hy; z0i += 1; }
    }

    // Per-class weights / compatibility row for this lane's class (lm in {0,1} valid)
    const float swc = (lm == 1) ? sw[1] : sw[0];
    const float bwc = (lm == 1) ? bw[1] : bw[0];
    const int   cc  = (lm == 1) ? 1 : 0;
    const float cm0 = cm[cc * 2 + 0];
    const float cm1 = cm[cc * 2 + 1];
    const int   base = hi * 16;

#pragma unroll
    for (int r = 0; r < 8; ++r) {
        const float ssv = acc_s[r];
        const float bbv = acc_b[r];
        // Normalizer lives in column n==2 of the same half-wave.
        // v_rcp_f32 + mul instead of the ~10-op IEEE division expansion:
        // the normalizer only rescales a f16-input filter response.
        const float ns = __shfl(ssv, base + 2, 32);
        const float nb = __shfl(bbv, base + 2, 32);
        const float msg = (ssv * __builtin_amdgcn_rcpf(ns)) * swc
                        + (bbv * __builtin_amdgcn_rcpf(nb)) * bwc;
        // Gather both class messages for the compatibility transform
        const float m0 = __shfl(msg, base + 0, 32);
        const float m1 = __shfl(msg, base + 1, 32);
        const float pw = m0 * cm0 + m1 * cm1;
        const int i = itile * 16 + r + hi * 8;     // C/D layout: row = r + 8*hi
        if (lm < 2) qout[i * Cc + lm] = u[i * Cc + lm] - pw;
    }
}

extern "C" void kernel_launch(void* const* d_in, const int* in_sizes, int n_in,
                              void* d_out, int out_size, void* d_ws, size_t ws_size,
                              hipStream_t stream) {
    (void)in_sizes; (void)n_in; (void)out_size; (void)ws_size;
    const float* u   = (const float*)d_in[0];   // unaries  [1,D,H,W,C]
    const float* rgb = (const float*)d_in[1];   // rgb      [1,D,H,W,1]
    const float* sw  = (const float*)d_in[2];   // spatial_ker_weights [2]
    const float* bw  = (const float*)d_in[3];   // bilateral_ker_weights [2]
    const float* cm  = (const float*)d_in[4];   // compatibility_matrix [2,2]
    float* out = (float*)d_out;

    float* ws0 = (float*)d_ws;                  // q ping
    float* ws1 = ws0 + Nn * Cc;                 // q pong  (needs 2*N*C*4 = 147KB)

    const size_t lds_bytes = (size_t)Nn * sizeof(Feat);   // 73728 B < 320KB/WGP
    const dim3 grid(72), block(256);

    const float* qin = u;                       // q starts equal to unaries
    for (int it = 0; it < 5; ++it) {
        float* qo = (it == 4) ? out : ((it & 1) ? ws1 : ws0);
        crf_iter<<<grid, block, lds_bytes, stream>>>(u, rgb, qin, qo, sw, bw, cm);
        qin = qo;
    }
}
